// ForcedGPULSTM_53137335386703
// MI455X (gfx1250) — matrix operand order repaired and checked by reading, not compiled, not run on hardware
//
#include <hip/hip_runtime.h>

// ---------------------------------------------------------------------------
// 2-layer LSTM (B=32, T=2048, I=256, H=512) for MI455X / gfx1250.
//
//  * Batch rows independent -> 2 independent workgroups (16 rows each), no
//    grid sync; the whole T=2048 scan runs inside one kernel.
//  * Each of 16 waves owns two (i,f,g,o) column quadruples -> the LSTM cell
//    update is wave-local in the WMMA f32 accumulators; c-state lives in
//    VGPRs across all steps; h-state lives in LDS as bf16 (16 KB/layer).
//  * GEMMs use v_wmma_f32_16x16x32_bf16 (K=32, f32 accumulate).
//  * Weights are bf16 in d_ws (7 MB, L2-resident). A laundered zero offset
//    (opaque SGPR, refreshed per time step) is folded into every weight-load
//    index: blocks LICM from hoisting the loop-invariant B-fragment loads
//    (round-1 spill bug) while keeping global-addr-space provenance so loads
//    lower to global_load_b128 (saddr form), not flat_load (round-2 wart).
// ---------------------------------------------------------------------------

#define B_ 32
#define T_ 2048
#define I_ 256
#define H_ 512
#define NTHREADS 512   // 16 waves of 32

typedef __bf16    bf16x16 __attribute__((ext_vector_type(16)));
typedef float     f32x8   __attribute__((ext_vector_type(8)));
typedef unsigned  u32x4   __attribute__((ext_vector_type(4)));

union ABFrag { u32x4 q[2]; bf16x16 v; };

static __device__ __forceinline__ unsigned short f2bf(float f) {
  unsigned u = __builtin_bit_cast(unsigned, f);
  u += 0x7FFFu + ((u >> 16) & 1u);           // round-to-nearest-even
  return (unsigned short)(u >> 16);
}
static __device__ __forceinline__ float bf2f(unsigned short h) {
  unsigned u = (unsigned)h << 16;
  return __builtin_bit_cast(float, u);
}
static __device__ __forceinline__ float sigm(float x) {
  return 1.f / (1.f + __expf(-x));
}
static __device__ __forceinline__ float tanh_fast(float x) {
  float t = __expf(-2.f * fabsf(x));
  float r = (1.f - t) / (1.f + t);
  return copysignf(r, x);
}

// A fragment (16x32 bf16, row-major in LDS, row length row_len halves).
// ISA layout: lane<16 -> M=lane, K in {kb..kb+7, kb+16..kb+23};
//             lane>=16 -> M=lane-16, K shifted by +8.
static __device__ __forceinline__ bf16x16
load_a_lds(const unsigned short* s, int row_len, int lane, int kbase) {
  int m  = lane & 15;
  int kh = (lane >> 4) & 1;
  const unsigned short* p = s + m * row_len + kbase + kh * 8;
  ABFrag a;
  a.q[0] = *(const u32x4*)(p);        // K = kb+kh*8 .. +7
  a.q[1] = *(const u32x4*)(p + 16);   // K = kb+16+kh*8 .. +7
  return a.v;
}

// B fragment (32x16 bf16). B[k][n] = W[n][k] (gates = h @ W^T), W row-major
// with row length ldk. ISA layout: lane<16 -> N=lane, K in {kb..kb+7, kb+16..kb+23};
// lane>=16 -> K shifted by +8.
static __device__ __forceinline__ bf16x16
load_b_glb(const unsigned short* __restrict__ w, int ldk, int lane, int nrow,
           int kbase) {
  const unsigned short* p =
      w + (size_t)nrow * ldk + kbase + ((lane >> 4) & 1) * 8;
  ABFrag b;
  b.q[0] = *(const u32x4*)(p);
  b.q[1] = *(const u32x4*)(p + 16);
  return b.v;
}

__global__ void cvt_f32_bf16(const float* __restrict__ s,
                             unsigned short* __restrict__ d, int n) {
  int i = blockIdx.x * blockDim.x + threadIdx.x;
  if (i < n) d[i] = f2bf(s[i]);
}

__global__ __launch_bounds__(NTHREADS, 1)
void lstm2_persistent(const float* __restrict__ x,
                      const unsigned short* __restrict__ w0_ih,
                      const unsigned short* __restrict__ w0_hh,
                      const float* __restrict__ b0,
                      const unsigned short* __restrict__ w1_ih,
                      const unsigned short* __restrict__ w1_hh,
                      const float* __restrict__ b1,
                      float* __restrict__ out) {
  __shared__ unsigned short xs[16 * I_];    // x_t tile, bf16   (8 KB)
  __shared__ unsigned short h0s[16 * H_];   // layer-0 h, bf16  (16 KB)
  __shared__ unsigned short h1s[16 * H_];   // layer-1 h, bf16  (16 KB)

  const int tid  = threadIdx.x;
  const int wave = tid >> 5;
  const int lane = tid & 31;
  const int b0r  = blockIdx.x * 16;          // batch-row base (0 or 16)
  const int nl   = lane & 15;
  const int mrow = (lane >> 4) ? 8 : 0;      // D-tile: row = r + mrow

  // Quadruple p in {0,1}: h columns wave*16 / (wave+16)*16.
  // Gate g in {0..3}: W row (gate column) = hcol[p] + g*512.
  int hcol[2];
  hcol[0] = wave * 16 + nl;
  hcol[1] = (wave + 16) * 16 + nl;

  float bias0v[8], bias1v[8];
#pragma unroll
  for (int p = 0; p < 2; ++p)
#pragma unroll
    for (int g = 0; g < 4; ++g) {
      bias0v[p * 4 + g] = b0[hcol[p] + g * H_];
      bias1v[p * 4 + g] = b1[hcol[p] + g * H_];
    }

  for (int e = tid; e < 16 * H_; e += NTHREADS) { h0s[e] = 0; h1s[e] = 0; }
  f32x8 c0st[2], c1st[2];
#pragma unroll
  for (int p = 0; p < 2; ++p)
#pragma unroll
    for (int r = 0; r < 8; ++r) { c0st[p][r] = 0.f; c1st[p][r] = 0.f; }
  __syncthreads();

#pragma unroll 1
  for (int t = 0; t < T_; ++t) {
    // Opaque zero, refreshed each step and folded into every weight-load
    // index: the address is no longer provably loop-invariant (no LICM /
    // cross-iteration CSE), but pointer provenance stays global.
    int lic = 0;
    asm volatile("" : "+s"(lic));

    // ---- stage x_t -> LDS (f32 -> bf16), coalesced ----
    const float* xt = x + (size_t)t * I_;
#pragma unroll 1
    for (int e = tid; e < 16 * I_; e += NTHREADS) {
      int bb = e >> 8, ii = e & (I_ - 1);
      xs[e] = f2bf(xt[(size_t)(b0r + bb) * T_ * I_ + ii]);
    }
    __syncthreads();   // xs ready; also orders prev-step h1s writes

    // ---- layer 0 gates: x_t @ w0_ih^T + h0 @ w0_hh^T + b0 ----
    f32x8 acc[8];
#pragma unroll
    for (int q = 0; q < 8; ++q)
#pragma unroll
      for (int r = 0; r < 8; ++r) acc[q][r] = bias0v[q];

#pragma unroll 1
    for (int kc = 0; kc < I_; kc += 32) {
      bf16x16 a = load_a_lds(xs, I_, lane, kc);
#pragma unroll
      for (int p = 0; p < 2; ++p)
#pragma unroll
        for (int g = 0; g < 4; ++g) {
          bf16x16 b =
              load_b_glb(w0_ih, I_, lane, hcol[p] + g * H_, kc + lic);
          acc[p * 4 + g] = __builtin_amdgcn_wmma_f32_16x16x32_bf16(
              false, a, false, b, (short)0, acc[p * 4 + g], false, false);
        }
    }
#pragma unroll 1
    for (int kc = 0; kc < H_; kc += 32) {
      bf16x16 a = load_a_lds(h0s, H_, lane, kc);
#pragma unroll
      for (int p = 0; p < 2; ++p)
#pragma unroll
        for (int g = 0; g < 4; ++g) {
          bf16x16 b =
              load_b_glb(w0_hh, H_, lane, hcol[p] + g * H_, kc + lic);
          acc[p * 4 + g] = __builtin_amdgcn_wmma_f32_16x16x32_bf16(
              false, a, false, b, (short)0, acc[p * 4 + g], false, false);
        }
    }
    __syncthreads();   // all waves done reading h0s

    // ---- layer 0 cell update (wave-local: i,f,g,o all in this wave) ----
#pragma unroll
    for (int p = 0; p < 2; ++p)
#pragma unroll
      for (int r = 0; r < 8; ++r) {
        float iv = sigm(acc[p * 4 + 0][r]);
        float fv = sigm(acc[p * 4 + 1][r]);
        float gv = tanh_fast(acc[p * 4 + 2][r]);
        float ov = sigm(acc[p * 4 + 3][r]);
        float cv = fv * c0st[p][r] + iv * gv;
        c0st[p][r] = cv;
        h0s[(r + mrow) * H_ + hcol[p]] = f2bf(ov * tanh_fast(cv));
      }
    __syncthreads();   // h0s ready

    // ---- layer 1 gates: h0 @ w1_ih^T + h1 @ w1_hh^T + b1 ----
#pragma unroll
    for (int q = 0; q < 8; ++q)
#pragma unroll
      for (int r = 0; r < 8; ++r) acc[q][r] = bias1v[q];

#pragma unroll 1
    for (int kc = 0; kc < H_; kc += 32) {
      bf16x16 a = load_a_lds(h0s, H_, lane, kc);
#pragma unroll
      for (int p = 0; p < 2; ++p)
#pragma unroll
        for (int g = 0; g < 4; ++g) {
          bf16x16 b =
              load_b_glb(w1_ih, H_, lane, hcol[p] + g * H_, kc + lic);
          acc[p * 4 + g] = __builtin_amdgcn_wmma_f32_16x16x32_bf16(
              false, a, false, b, (short)0, acc[p * 4 + g], false, false);
        }
    }
#pragma unroll 1
    for (int kc = 0; kc < H_; kc += 32) {
      bf16x16 a = load_a_lds(h1s, H_, lane, kc);
#pragma unroll
      for (int p = 0; p < 2; ++p)
#pragma unroll
        for (int g = 0; g < 4; ++g) {
          bf16x16 b =
              load_b_glb(w1_hh, H_, lane, hcol[p] + g * H_, kc + lic);
          acc[p * 4 + g] = __builtin_amdgcn_wmma_f32_16x16x32_bf16(
              false, a, false, b, (short)0, acc[p * 4 + g], false, false);
        }
    }
    __syncthreads();   // all waves done reading h1s

    // ---- layer 1 cell update + sequence output ----
#pragma unroll
    for (int p = 0; p < 2; ++p)
#pragma unroll
      for (int r = 0; r < 8; ++r) {
        float iv = sigm(acc[p * 4 + 0][r]);
        float fv = sigm(acc[p * 4 + 1][r]);
        float gv = tanh_fast(acc[p * 4 + 2][r]);
        float ov = sigm(acc[p * 4 + 3][r]);
        float cv = fv * c1st[p][r] + iv * gv;
        c1st[p][r] = cv;
        float hv = ov * tanh_fast(cv);
        h1s[(r + mrow) * H_ + hcol[p]] = f2bf(hv);
        out[((size_t)(b0r + r + mrow) * T_ + t) * H_ + hcol[p]] = hv;
      }
    // next-iter staging barrier orders h1s writes vs next gemm1 reads
  }

  __syncthreads();
  // ---- final states: h_n [L,B,H] then c_n [L,B,H] ----
  const size_t OUT0 = (size_t)B_ * T_ * H_;
  for (int e = tid; e < 16 * H_; e += NTHREADS) {
    int bb = e >> 9, cc = e & (H_ - 1);
    out[OUT0 + (size_t)(b0r + bb) * H_ + cc]      = bf2f(h0s[e]);
    out[OUT0 + (size_t)(B_ + b0r + bb) * H_ + cc] = bf2f(h1s[e]);
  }
  const size_t OUTC = OUT0 + 2ull * B_ * H_;
#pragma unroll
  for (int p = 0; p < 2; ++p)
#pragma unroll
    for (int r = 0; r < 8; ++r) {
      out[OUTC + (size_t)(b0r + r + mrow) * H_ + hcol[p]]      = c0st[p][r];
      out[OUTC + (size_t)(B_ + b0r + r + mrow) * H_ + hcol[p]] = c1st[p][r];
    }
}

extern "C" void kernel_launch(void* const* d_in, const int* in_sizes, int n_in,
                              void* d_out, int out_size, void* d_ws,
                              size_t ws_size, hipStream_t stream) {
  const float* x     = (const float*)d_in[0];
  const float* w0_ih = (const float*)d_in[1];
  const float* w0_hh = (const float*)d_in[2];
  const float* b0    = (const float*)d_in[3];
  const float* w1_ih = (const float*)d_in[4];
  const float* w1_hh = (const float*)d_in[5];
  const float* b1    = (const float*)d_in[6];
  float*       out   = (float*)d_out;

  // bf16 weight staging in workspace (7 MB total, L2-resident)
  unsigned short* wb0_ih = (unsigned short*)d_ws;
  unsigned short* wb0_hh = wb0_ih + 4 * H_ * I_;   // 2048*256
  unsigned short* wb1_ih = wb0_hh + 4 * H_ * H_;   // 2048*512
  unsigned short* wb1_hh = wb1_ih + 4 * H_ * H_;

  const int n_ih0 = 4 * H_ * I_;   // 524288
  const int n_hh  = 4 * H_ * H_;   // 1048576
  cvt_f32_bf16<<<(n_ih0 + 255) / 256, 256, 0, stream>>>(w0_ih, wb0_ih, n_ih0);
  cvt_f32_bf16<<<(n_hh  + 255) / 256, 256, 0, stream>>>(w0_hh, wb0_hh, n_hh);
  cvt_f32_bf16<<<(n_hh  + 255) / 256, 256, 0, stream>>>(w1_ih, wb1_ih, n_hh);
  cvt_f32_bf16<<<(n_hh  + 255) / 256, 256, 0, stream>>>(w1_hh, wb1_hh, n_hh);

  // 2 independent workgroups (one per 16-row batch half), 16 waves each.
  lstm2_persistent<<<2, NTHREADS, 0, stream>>>(x, wb0_ih, wb0_hh, b0, wb1_ih,
                                               wb1_hh, b1, out);
}
